// chan_selfattn_37134287241224
// MI455X (gfx1250) — matrix operand order repaired and checked by reading, not compiled, not run on hardware
//
#include <hip/hip_runtime.h>

// ---------------------------------------------------------------------------
// Channel self-attention (XCA) for MI455X / gfx1250, wave32 + WMMA bf16.
//
// Pipeline (bandwidth-minimizing, bf16 intermediates):
//   1. x (f32, [B,C,N]) -> xT bf16 [B,N,C]  (tiled LDS transpose)
//   2. qkv GEMM:  qkv_w(1536x512) @ x(512x4096)  via v_wmma_f32_16x16x32_bf16
//   3. depthwise k=3 conv along N (+bias); side-effect: sum(q^2),sum(k^2)
//   4. per (b,h): 64x64 Gram S = q_raw @ k_raw^T (K=4096, WMMA);
//      epilogue applies temperature / (||q_c|| ||k_d||) and softmax in LDS
//   5. fused matrix M_b = proj_w @ blockdiag(attn_b)  (tiny)
//   6. final GEMM: out = M_b(512x512) @ v(512x4096) + proj_b  (WMMA, f32 out)
//
// Round-1 fix: GEMM waves were spilling accumulators (compiler targeted
// ~128 VGPRs/wave). Now: waves_per_eu(2) => ~512 VGPR budget, single
// base pointer + 24-bit immediate offsets per operand, B-then-A MAC order.
// ---------------------------------------------------------------------------

typedef __bf16 bf16_t;
typedef __attribute__((ext_vector_type(16))) __bf16 v16bf;
typedef __attribute__((ext_vector_type(8)))  __bf16 v8bf;
typedef __attribute__((ext_vector_type(8)))  float  v8f;

#define DIMC   512
#define HEADS  8
#define BATCH  8
#define NN     4096
#define D3     1536
#define CH     64
#define EPSN   1e-12f

// -------------------- WMMA fragment helpers (ISA 7.12.2 layouts) -----------

// A fragment (16x32 bf16, M x K): lane half = lane>>4, m = lane&15.
//   elements 0..7  -> K = k0 + 8*half + i      (8*half folded into pointer)
//   elements 8..15 -> K = k0 + 16 + 8*half + (i-8)
// p points at (row m, k0 + 8*half); lo at +0, hi at +16 elements.
static __device__ __forceinline__ v16bf make_a_frag(const bf16_t* p) {
    const v8bf lo = *(const v8bf*)(p);
    const v8bf hi = *(const v8bf*)(p + 16);
    v16bf a;
#pragma unroll
    for (int i = 0; i < 8; ++i) { a[i] = lo[i]; a[i + 8] = hi[i]; }
    return a;
}

static __device__ __forceinline__ v8f wmma_bf16(v16bf a, v16bf b, v8f c) {
    return __builtin_amdgcn_wmma_f32_16x16x32_bf16(
        /*neg_a=*/false, a, /*neg_b=*/false, b,
        /*c_mod=*/(short)0, c, /*reuse_a=*/false, /*reuse_b=*/false);
}

// -------------------- 1. transpose + cast x -> bf16 [B,N,C] ----------------

__global__ void k_transpose_x(const float* __restrict__ x, bf16_t* __restrict__ xT) {
    __shared__ float tile[32][33];
    const int b  = blockIdx.z;
    const int n0 = blockIdx.x * 32;
    const int c0 = blockIdx.y * 32;
#pragma unroll
    for (int j = 0; j < 32; j += 8) {
        int c = c0 + threadIdx.y + j;
        int n = n0 + threadIdx.x;
        tile[threadIdx.y + j][threadIdx.x] = x[((long)b * DIMC + c) * NN + n];
    }
    __syncthreads();
#pragma unroll
    for (int j = 0; j < 32; j += 8) {
        int n = n0 + threadIdx.y + j;
        int c = c0 + threadIdx.x;
        xT[((long)b * NN + n) * DIMC + c] = (bf16_t)tile[threadIdx.x][threadIdx.y + j];
    }
}

__global__ void k_cast_bf16(const float* __restrict__ s, bf16_t* __restrict__ d, int n) {
    int i = blockIdx.x * blockDim.x + threadIdx.x;
    if (i < n) d[i] = (bf16_t)s[i];
}

__global__ void k_zero_f32(float* __restrict__ p, int n) {
    int i = blockIdx.x * blockDim.x + threadIdx.x;
    if (i < n) p[i] = 0.0f;
}

// -------------------- 2. qkv pointwise GEMM (WMMA bf16) --------------------
// D[o,n] = sum_c W[o,c] * x[c,n] + bias[o]; per-batch B operand = xT[b][n][c].
// 8 waves/block, wave tile 64x64 (4x4 wmma), block tile 256x128.

__global__ __launch_bounds__(256)
__attribute__((amdgpu_waves_per_eu(2)))
void k_gemm_qkv(const bf16_t* __restrict__ Wbf,  // [D3, DIMC]
                const bf16_t* __restrict__ xT,   // [B, NN, DIMC]
                const float*  __restrict__ bias, // [D3]
                bf16_t* __restrict__ qkv)        // [B, D3, NN]
{
    const int b     = blockIdx.z;
    const int tileM = blockIdx.y * 256;
    const int tileN = blockIdx.x * 128;
    const int wid   = threadIdx.x >> 5;
    const int lane  = threadIdx.x & 31;
    const int half  = lane >> 4, l16 = lane & 15;
    const int waveM = tileM + (wid >> 1) * 64;
    const int waveN = tileN + (wid & 1) * 64;

    // single base pointer per operand; mt/nt tile offsets become immediates
    const bf16_t* aptr = Wbf + (long)(waveM + l16) * DIMC + 8 * half;
    const bf16_t* bptr = xT + (long)b * NN * DIMC
                            + (long)(waveN + l16) * DIMC + 8 * half;

    v8f acc[4][4];
    const v8f z = {0.f, 0.f, 0.f, 0.f, 0.f, 0.f, 0.f, 0.f};
#pragma unroll
    for (int i = 0; i < 4; ++i)
#pragma unroll
        for (int j = 0; j < 4; ++j) acc[i][j] = z;

    for (int k0 = 0; k0 < DIMC; k0 += 32) {
        v16bf bfm[4];
#pragma unroll
        for (int nt = 0; nt < 4; ++nt)
            bfm[nt] = make_a_frag(bptr + nt * 16 * DIMC);
#pragma unroll
        for (int mt = 0; mt < 4; ++mt) {
            const v16bf af = make_a_frag(aptr + mt * 16 * DIMC);
#pragma unroll
            for (int nt = 0; nt < 4; ++nt)
                acc[mt][nt] = wmma_bf16(af, bfm[nt], acc[mt][nt]);
        }
        aptr += 32;
        bptr += 32;
    }

#pragma unroll
    for (int mt = 0; mt < 4; ++mt)
#pragma unroll
        for (int nt = 0; nt < 4; ++nt)
#pragma unroll
            for (int r = 0; r < 8; ++r) {
                int m = waveM + mt * 16 + r + 8 * half;   // D: M = r + 8*half
                int n = waveN + nt * 16 + l16;            // D: N = lane&15
                float v = acc[mt][nt][r] + bias[m];
                qkv[((long)b * D3 + m) * NN + n] = (bf16_t)v;
            }
}

// -------------------- 3. depthwise conv + norm side-products ---------------
// q,k (c3 < 1024): write [b][c3][n] bf16 + accumulate sum(y^2) into norms.
// v   (c3 >=1024): write transposed vT[b][n][c3-1024] for the final GEMM.

__global__ void k_dwconv(const bf16_t* __restrict__ qkv,
                         const float*  __restrict__ dw_w,   // [D3,1,3]
                         const float*  __restrict__ dw_b,   // [D3]
                         bf16_t* __restrict__ qk,           // [B, 2*DIMC, NN]
                         bf16_t* __restrict__ vT,           // [B, NN, DIMC]
                         float*  __restrict__ norms)        // [B, 2*DIMC]
{
    __shared__ float red[256];
    const int b  = blockIdx.z;
    const int c3 = blockIdx.y;
    const int n  = blockIdx.x * 256 + threadIdx.x;
    const bf16_t* in = qkv + ((long)b * D3 + c3) * NN;

    const float w0 = dw_w[c3 * 3 + 0], w1 = dw_w[c3 * 3 + 1], w2 = dw_w[c3 * 3 + 2];
    const float xl = (n > 0)      ? (float)in[n - 1] : 0.0f;
    const float xc = (float)in[n];
    const float xr = (n < NN - 1) ? (float)in[n + 1] : 0.0f;
    const float y  = w0 * xl + w1 * xc + w2 * xr + dw_b[c3];

    if (c3 < 2 * DIMC) {
        qk[((long)b * 2 * DIMC + c3) * NN + n] = (bf16_t)y;
        red[threadIdx.x] = y * y;
        __syncthreads();
#pragma unroll
        for (int s = 128; s > 0; s >>= 1) {
            if (threadIdx.x < s) red[threadIdx.x] += red[threadIdx.x + s];
            __syncthreads();
        }
        if (threadIdx.x == 0) atomicAdd(&norms[b * 2 * DIMC + c3], red[0]);
    } else {
        vT[((long)b * NN + n) * DIMC + (c3 - 2 * DIMC)] = (bf16_t)y;
    }
}

// -------------------- 4. Gram (q @ k^T, K=4096) + scaled softmax ----------
// One workgroup (4 waves) per (b,h). L2 normalization is folded into the
// score epilogue: S'_{cd} = S_cd * temp_h / (max(||q_c||,eps)*max(||k_d||,eps)).

__global__ __launch_bounds__(128)
void k_attn(const bf16_t* __restrict__ qk,      // [B, 2*DIMC, NN]
            const float*  __restrict__ norms,   // [B, 2*DIMC] sumsq
            const float*  __restrict__ temperature, // [HEADS]
            float* __restrict__ attn)           // [B, HEADS, CH, CH]
{
    __shared__ float S[64][65];
    __shared__ float rnkS[64];
    const int h    = blockIdx.x, b = blockIdx.y;
    const int tid  = threadIdx.x;                // 128 threads = 4 waves
    const int wid  = tid >> 5, lane = tid & 31;
    const int half = lane >> 4, l16 = lane & 15;

    // A operand: q rows (ci), K along n (contiguous).  8*half folded in.
    const bf16_t* aptr = qk + ((long)b * 2 * DIMC + h * CH + l16) * NN + 8 * half;
    // B operand: column d = wid*16 + l16 of k^T; K along n (contiguous).
    const bf16_t* bptr = qk + ((long)b * 2 * DIMC + DIMC + h * CH + wid * 16 + l16) * NN
                            + 8 * half;

    v8f acc[4];
    const v8f z = {0.f, 0.f, 0.f, 0.f, 0.f, 0.f, 0.f, 0.f};
#pragma unroll
    for (int i = 0; i < 4; ++i) acc[i] = z;

    for (int k0 = 0; k0 < NN; k0 += 32) {
        const v16bf bfm = make_a_frag(bptr);
#pragma unroll
        for (int mt = 0; mt < 4; ++mt) {
            const v16bf af = make_a_frag(aptr + mt * 16 * NN);
            acc[mt] = wmma_bf16(af, bfm, acc[mt]);
        }
        aptr += 32;
        bptr += 32;
    }

#pragma unroll
    for (int mt = 0; mt < 4; ++mt)
#pragma unroll
        for (int r = 0; r < 8; ++r)
            S[mt * 16 + r + 8 * half][wid * 16 + l16] = acc[mt][r];

    if (tid < 64) {
        float nk = norms[b * 2 * DIMC + DIMC + h * CH + tid];
        rnkS[tid] = 1.0f / fmaxf(sqrtf(nk), EPSN);
    }
    __syncthreads();

    if (tid < 64) {
        const int row = tid;
        const float rq = 1.0f / fmaxf(sqrtf(norms[b * 2 * DIMC + h * CH + row]), EPSN);
        const float t  = temperature[h];
        float vbuf[64];
        float mx = -3.0e38f;
#pragma unroll
        for (int d = 0; d < 64; ++d) {
            float s = S[row][d] * t * rq * rnkS[d];
            vbuf[d] = s;
            mx = fmaxf(mx, s);
        }
        float sum = 0.0f;
#pragma unroll
        for (int d = 0; d < 64; ++d) {
            float e = __expf(vbuf[d] - mx);
            vbuf[d] = e;
            sum += e;
        }
        const float inv = 1.0f / sum;
        float* ap = attn + ((long)(b * HEADS + h) * CH + row) * CH;
#pragma unroll
        for (int d = 0; d < 64; ++d) ap[d] = vbuf[d] * inv;
    }
}

// -------------------- 5. fused matrix M_b = proj_w @ blockdiag(attn_b) ----
// M[b][o][h*64+d] = sum_ci proj_w[o][h*64+ci] * attn[b][h][ci][d]

__global__ void k_build_m(const float* __restrict__ proj_w,  // [DIMC, DIMC]
                          const float* __restrict__ attn,    // [B,HEADS,CH,CH]
                          bf16_t* __restrict__ Mbf)          // [B, DIMC, DIMC]
{
    const int b   = blockIdx.y;
    const int idx = blockIdx.x * 256 + threadIdx.x;  // DIMC*DIMC per batch
    const int o   = idx >> 9;
    const int cp  = idx & 511;
    const int h   = cp >> 6, d = cp & 63;
    const float* pw = proj_w + (long)o * DIMC + h * CH;
    const float* at = attn + ((long)(b * HEADS + h) * CH) * CH + d;
    float s = 0.0f;
#pragma unroll
    for (int ci = 0; ci < CH; ++ci) s += pw[ci] * at[ci * CH];
    Mbf[((long)b * DIMC + o) * DIMC + cp] = (bf16_t)s;
}

// -------------------- 6. final fused GEMM: out = M_b @ v + proj_b ---------

__global__ __launch_bounds__(256)
__attribute__((amdgpu_waves_per_eu(2)))
void k_gemm_out(const bf16_t* __restrict__ Mbf,   // [B, DIMC, DIMC]
                const bf16_t* __restrict__ vT,    // [B, NN, DIMC]
                const float*  __restrict__ proj_b,// [DIMC]
                float* __restrict__ out)          // [B, DIMC, NN]
{
    const int b     = blockIdx.z;
    const int tileM = blockIdx.y * 256;
    const int tileN = blockIdx.x * 128;
    const int wid   = threadIdx.x >> 5;
    const int lane  = threadIdx.x & 31;
    const int half  = lane >> 4, l16 = lane & 15;
    const int waveM = tileM + (wid >> 1) * 64;
    const int waveN = tileN + (wid & 1) * 64;

    const bf16_t* aptr = Mbf + (long)b * DIMC * DIMC
                             + (long)(waveM + l16) * DIMC + 8 * half;
    const bf16_t* bptr = vT + (long)b * NN * DIMC
                            + (long)(waveN + l16) * DIMC + 8 * half;

    v8f acc[4][4];
    const v8f z = {0.f, 0.f, 0.f, 0.f, 0.f, 0.f, 0.f, 0.f};
#pragma unroll
    for (int i = 0; i < 4; ++i)
#pragma unroll
        for (int j = 0; j < 4; ++j) acc[i][j] = z;

    for (int k0 = 0; k0 < DIMC; k0 += 32) {
        v16bf bfm[4];
#pragma unroll
        for (int nt = 0; nt < 4; ++nt)
            bfm[nt] = make_a_frag(bptr + nt * 16 * DIMC);
#pragma unroll
        for (int mt = 0; mt < 4; ++mt) {
            const v16bf af = make_a_frag(aptr + mt * 16 * DIMC);
#pragma unroll
            for (int nt = 0; nt < 4; ++nt)
                acc[mt][nt] = wmma_bf16(af, bfm[nt], acc[mt][nt]);
        }
        aptr += 32;
        bptr += 32;
    }

#pragma unroll
    for (int mt = 0; mt < 4; ++mt)
#pragma unroll
        for (int nt = 0; nt < 4; ++nt)
#pragma unroll
            for (int r = 0; r < 8; ++r) {
                int m = waveM + mt * 16 + r + 8 * half;
                int n = waveN + nt * 16 + l16;
                out[((long)b * DIMC + m) * NN + n] = acc[mt][nt][r] + proj_b[m];
            }
}

// -------------------- launch ----------------------------------------------

extern "C" void kernel_launch(void* const* d_in, const int* in_sizes, int n_in,
                              void* d_out, int out_size, void* d_ws, size_t ws_size,
                              hipStream_t stream) {
    const float* x      = (const float*)d_in[0];
    const float* qkv_w  = (const float*)d_in[1];
    const float* qkv_b  = (const float*)d_in[2];
    const float* dw_w   = (const float*)d_in[3];
    const float* dw_b   = (const float*)d_in[4];
    const float* proj_w = (const float*)d_in[5];
    const float* proj_b = (const float*)d_in[6];
    const float* temp   = (const float*)d_in[7];

    char* ws = (char*)d_ws;
    size_t off = 0;
    bf16_t* xT    = (bf16_t*)(ws + off); off += (size_t)BATCH * NN * DIMC * 2;   // 32 MB
    bf16_t* Wbf   = (bf16_t*)(ws + off); off += (size_t)D3 * DIMC * 2;           // 1.5 MB
    bf16_t* qkv   = (bf16_t*)(ws + off); off += (size_t)BATCH * D3 * NN * 2;     // 96 MB
    bf16_t* qk    = (bf16_t*)(ws + off); off += (size_t)BATCH * 2 * DIMC * NN * 2;// 64 MB
    bf16_t* vTb   = (bf16_t*)(ws + off); off += (size_t)BATCH * NN * DIMC * 2;   // 32 MB
    float*  norms = (float*) (ws + off); off += (size_t)BATCH * 2 * DIMC * 4;    // 32 KB
    float*  attn  = (float*) (ws + off); off += (size_t)BATCH * HEADS * CH * CH * 4; // 1 MB
    bf16_t* Mbf   = (bf16_t*)(ws + off); off += (size_t)BATCH * DIMC * DIMC * 2; // 4 MB
    float*  out   = (float*)d_out;

    k_transpose_x<<<dim3(NN / 32, DIMC / 32, BATCH), dim3(32, 8), 0, stream>>>(x, xT);
    k_cast_bf16<<<(D3 * DIMC + 255) / 256, 256, 0, stream>>>(qkv_w, Wbf, D3 * DIMC);
    k_zero_f32<<<(BATCH * 2 * DIMC + 255) / 256, 256, 0, stream>>>(norms, BATCH * 2 * DIMC);
    k_gemm_qkv<<<dim3(NN / 128, D3 / 256, BATCH), 256, 0, stream>>>(Wbf, xT, qkv_b, qkv);
    k_dwconv<<<dim3(NN / 256, D3, BATCH), 256, 0, stream>>>(qkv, dw_w, dw_b, qk, vTb, norms);
    k_attn<<<dim3(HEADS, BATCH), 128, 0, stream>>>(qk, norms, temp, attn);
    k_build_m<<<dim3(DIMC * DIMC / 256, BATCH), 256, 0, stream>>>(proj_w, attn, Mbf);
    k_gemm_out<<<dim3(NN / 128, DIMC / 256, BATCH), 256, 0, stream>>>(Mbf, vTb, proj_b, out);
}
